// STGN_LSTM_65592740545327
// MI455X (gfx1250) — matrix operand order repaired and checked by reading, compiled, not run on hardware
//
#include <hip/hip_runtime.h>

#define B_TOTAL 524288
#define SEQ_LEN 5
#define HID     64
#define NFRAG   36                      // 3 gates * 3 ktiles * 4 mtiles
#define APACK_HALVES (NFRAG*32*16)      // 18432 halves = 36 KB

typedef __attribute__((ext_vector_type(16))) _Float16 v16h;
typedef __attribute__((ext_vector_type(8)))  float    v8f;

// ---------------------------------------------------------------------------
// Prep kernel: pack augmented weight matrices [Wh | Wx | Wt | bias | 0-pad]
// (16x32 f16 A-fragments in WMMA lane layout) into d_ws.
// A layout (16-bit A 16x32): lane L (g=L/16, M=L%16), half j ->
//   K = 32*kt + (j/8)*16 + g*8 + (j%8),  row n = 16*mt + M.
// ---------------------------------------------------------------------------
__global__ void stgn_pack_weights(
    const float* __restrict__ Wfh_w, const float* __restrict__ Wfh_b,
    const float* __restrict__ Wfx_w, const float* __restrict__ Wfx_b,
    const float* __restrict__ bf,
    const float* __restrict__ Wih_w, const float* __restrict__ Wih_b,
    const float* __restrict__ Wix_w, const float* __restrict__ Wix_b,
    const float* __restrict__ bi,
    const float* __restrict__ WTh_w, const float* __restrict__ WTh_b,
    const float* __restrict__ WTx_w, const float* __restrict__ WTx_b,
    const float* __restrict__ WTt_w, const float* __restrict__ WTt_b,
    const float* __restrict__ bT,
    _Float16* __restrict__ Apack)
{
  const int frag = blockIdx.x;          // 0..35
  const int mt = frag & 3;
  const int kt = (frag >> 2) % 3;
  const int G  = frag / 12;             // 0=f, 1=i, 2=T
  const int lane = threadIdx.x;         // 0..31
  const int g = lane >> 4;
  const int n = mt*16 + (lane & 15);

  const float* Wh = (G==0) ? Wfh_w : (G==1) ? Wih_w : WTh_w;
  const float* Wx = (G==0) ? Wfx_w : (G==1) ? Wix_w : WTx_w;
  float bias;
  if (G == 0)      bias = Wfh_b[n] + Wfx_b[n] + bf[n];
  else if (G == 1) bias = Wih_b[n] + Wix_b[n] + bi[n];
  else             bias = WTh_b[n] + WTx_b[n] + WTt_b[n] + bT[n];

  _Float16* dst = Apack + ((size_t)frag*32 + lane)*16;
  #pragma unroll
  for (int j = 0; j < 16; ++j) {
    const int K = 32*kt + (j >> 3)*16 + g*8 + (j & 7);
    float v = 0.f;
    if (K < 64)       v = Wh[n*64 + K];
    else if (K == 64) v = Wx[n*2 + 0];
    else if (K == 65) v = Wx[n*2 + 1];
    else if (K == 66) v = (G == 2) ? WTt_w[n] : 0.f;
    else if (K == 67) v = bias;
    dst[j] = (_Float16)v;
  }
}

// ---------------------------------------------------------------------------
// Main kernel
// ---------------------------------------------------------------------------
__device__ __forceinline__ float sig_(float x)  { return 1.f / (1.f + __expf(-x)); }
__device__ __forceinline__ float tanh_(float x) { return 2.f / (1.f + __expf(-2.f*x)) - 1.f; }
// SWAPX16: lane L <-> lane L^16 (ds_swizzle bitmask: and=0x1f, or=0, xor=0x10)
__device__ __forceinline__ float swap16_(float x) {
  return __int_as_float(__builtin_amdgcn_ds_swizzle(__float_as_int(x), 0x401f));
}

// 12 WMMAs for one gate: acc[mt] = A[G] (16x96 aug weights) x B (96 x 16 batch)
__device__ __forceinline__ void gate_preact(const _Float16* lA, int lane, int G,
                                            const v16h& Bf0, const v16h& Bf1,
                                            const v16h& Bf2, v8f acc[4])
{
  #pragma unroll
  for (int mt = 0; mt < 4; ++mt)
    #pragma unroll
    for (int v = 0; v < 8; ++v) acc[mt][v] = 0.f;

  #pragma unroll
  for (int kt = 0; kt < 3; ++kt) {
    // Barrier for memory ops only: keeps the loop-invariant LDS fragment loads
    // from being hoisted out of the timestep loop (which spills to scratch).
    asm volatile("" ::: "memory");
    const v16h Bsel = (kt == 0) ? Bf0 : (kt == 1) ? Bf1 : Bf2;
    const _Float16* base = lA + (size_t)((G*3 + kt)*4) * 32 * 16;
    #pragma unroll
    for (int mt = 0; mt < 4; ++mt) {
      v16h a = *(const v16h*)(base + (mt*32 + lane)*16);
      acc[mt] = __builtin_amdgcn_wmma_f32_16x16x32_f16(
          false, a, false, Bsel, (short)0, acc[mt], false, false);
    }
  }
}

__global__ __launch_bounds__(256)
void stgn_main(const float* __restrict__ X,
               const _Float16* __restrict__ Apack,
               const float* __restrict__ cls_w,
               const float* __restrict__ cls_b,
               float* __restrict__ out)
{
  __shared__ __align__(32) _Float16 lA[APACK_HALVES];
  {
    const float4* s = (const float4*)Apack;
    float4* d = (float4*)lA;
    for (int i = threadIdx.x; i < APACK_HALVES/8; i += 256) d[i] = s[i];
  }
  __syncthreads();

  const int  lane  = threadIdx.x & 31;
  const int  wave  = threadIdx.x >> 5;
  const int  g     = lane >> 4;
  const int  col   = lane & 15;
  const long b0    = ((long)blockIdx.x * 8 + wave) * 16;
  const long batch = b0 + col;
  const _Float16 hz = (_Float16)0.f;

  // B-fragments of the state vector (K x 16 batch cols): kt0/kt1 = h (zero at t=0)
  v16h Bf0, Bf1;
  #pragma unroll
  for (int j = 0; j < 16; ++j) { Bf0[j] = hz; Bf1[j] = hz; }

  // cell state, transposed D layout: c[mt][v] -> H row 16*mt+v+8*g, batch col
  float c[4][8];
  #pragma unroll
  for (int mt = 0; mt < 4; ++mt)
    #pragma unroll
    for (int v = 0; v < 8; ++v) c[mt][v] = 0.f;

  for (int t = 0; t < SEQ_LEN; ++t) {
    const float* xr = X + ((size_t)batch * SEQ_LEN + t) * 3;
    const float x0 = xr[0], x1 = xr[1], dt = xr[2];

    // kt=2 B-fragment: K=64..95 -> [x0, x1, dt, 1, 0...] on half-wave 0, zeros on half 1
    v16h Bf2;
    #pragma unroll
    for (int j = 0; j < 16; ++j) Bf2[j] = hz;
    Bf2[0] = g ? hz : (_Float16)x0;
    Bf2[1] = g ? hz : (_Float16)x1;
    Bf2[2] = g ? hz : (_Float16)dt;
    Bf2[3] = g ? hz : (_Float16)1.0f;

    v8f acc[4];

    // Gate T: keep only tanh(sigmoid(preT) * dt)  (32 regs)
    float tTdt[4][8];
    gate_preact(lA, lane, 2, Bf0, Bf1, Bf2, acc);
    #pragma unroll
    for (int mt = 0; mt < 4; ++mt)
      #pragma unroll
      for (int v = 0; v < 8; ++v)
        tTdt[mt][v] = tanh_(sig_(acc[mt][v]) * dt);

    // Gate f: c = sigmoid(pref) * c   (in place)
    gate_preact(lA, lane, 0, Bf0, Bf1, Bf2, acc);
    #pragma unroll
    for (int mt = 0; mt < 4; ++mt)
      #pragma unroll
      for (int v = 0; v < 8; ++v)
        c[mt][v] = sig_(acc[mt][v]) * c[mt][v];

    // Gate i: c += sigmoid(prei) * tanh(T*dt)
    gate_preact(lA, lane, 1, Bf0, Bf1, Bf2, acc);
    #pragma unroll
    for (int mt = 0; mt < 4; ++mt)
      #pragma unroll
      for (int v = 0; v < 8; ++v)
        c[mt][v] += sig_(acc[mt][v]) * tTdt[mt][v];

    // h = tanh(c)
    float hD[4][8];
    #pragma unroll
    for (int mt = 0; mt < 4; ++mt)
      #pragma unroll
      for (int v = 0; v < 8; ++v)
        hD[mt][v] = tanh_(c[mt][v]);

    // Re-layout h (f32 D-layout) -> f16 B-fragments for next step (no lane exchange):
    // lane L half j   <- VGPR j of tile 2kt   held by lane L
    // lane L half j+8 <- VGPR j of tile 2kt+1 held by lane L
    #pragma unroll
    for (int kt = 0; kt < 2; ++kt) {
      v16h nb;
      #pragma unroll
      for (int j = 0; j < 8; ++j) {
        const float a0 = hD[2*kt    ][j];
        const float a1 = hD[2*kt + 1][j];
        nb[j]     = (_Float16)a0;
        nb[j + 8] = (_Float16)a1;
      }
      if (kt == 0) Bf0 = nb; else Bf1 = nb;
    }

    if (t == SEQ_LEN - 1) {
      // out[b] = sigmoid(sum_n h[n][b]*cls_w[n] + cls_b): per-lane partial + SWAPX16 reduce
      float part = 0.f;
      #pragma unroll
      for (int mt = 0; mt < 4; ++mt)
        #pragma unroll
        for (int v = 0; v < 8; ++v)
          part += hD[mt][v] * cls_w[16*mt + v + 8*g];
      const float tot = part + swap16_(part);
      if (g == 0) out[batch] = sig_(tot + cls_b[0]);
    }
  }
}

// ---------------------------------------------------------------------------
// Launch
// ---------------------------------------------------------------------------
extern "C" void kernel_launch(void* const* d_in, const int* in_sizes, int n_in,
                              void* d_out, int out_size, void* d_ws, size_t ws_size,
                              hipStream_t stream) {
  const float* X     = (const float*)d_in[0];
  const float* Wfh_w = (const float*)d_in[1];  const float* Wfh_b = (const float*)d_in[2];
  const float* Wfx_w = (const float*)d_in[3];  const float* Wfx_b = (const float*)d_in[4];
  const float* bf    = (const float*)d_in[5];
  const float* Wih_w = (const float*)d_in[6];  const float* Wih_b = (const float*)d_in[7];
  const float* Wix_w = (const float*)d_in[8];  const float* Wix_b = (const float*)d_in[9];
  const float* bi    = (const float*)d_in[10];
  const float* WTh_w = (const float*)d_in[11]; const float* WTh_b = (const float*)d_in[12];
  const float* WTx_w = (const float*)d_in[13]; const float* WTx_b = (const float*)d_in[14];
  const float* WTt_w = (const float*)d_in[15]; const float* WTt_b = (const float*)d_in[16];
  const float* bT    = (const float*)d_in[17];
  const float* cls_w = (const float*)d_in[18]; const float* cls_b = (const float*)d_in[19];

  _Float16* Apack = (_Float16*)d_ws;   // 36 KB of packed A-fragments

  stgn_pack_weights<<<NFRAG, 32, 0, stream>>>(
      Wfh_w, Wfh_b, Wfx_w, Wfx_b, bf,
      Wih_w, Wih_b, Wix_w, Wix_b, bi,
      WTh_w, WTh_b, WTx_w, WTx_b, WTt_w, WTt_b, bT, Apack);

  const int blocks = B_TOTAL / (8 * 16);   // 8 waves/block, 16 batch rows/wave
  stgn_main<<<blocks, 256, 0, stream>>>(X, Apack, cls_w, cls_b, (float*)d_out);
}
